// Attention_44719199486644
// MI455X (gfx1250) — compile-verified, matrix operand order checked by reading, not run on hardware
//
#include <hip/hip_runtime.h>
#include <hip/hip_bf16.h>

// ---------- types ----------
typedef __attribute__((ext_vector_type(16))) __bf16         bf16x16;
typedef __attribute__((ext_vector_type(16))) unsigned short u16x16;
typedef __attribute__((ext_vector_type(8)))  float          f32x8;

static __device__ __forceinline__ f32x8 zero8() {
    f32x8 z = {0.f, 0.f, 0.f, 0.f, 0.f, 0.f, 0.f, 0.f};
    return z;
}

// f32 -> bf16 raw bits, round-to-nearest-even
static __device__ __forceinline__ unsigned short f2bf(float f) {
    unsigned int u = __builtin_bit_cast(unsigned int, f);
    u += 0x7FFFu + ((u >> 16) & 1u);
    return (unsigned short)(u >> 16);
}

// A-style fragment (ISA 7.12.2, 16-bit A 16x32): lane holds one row;
// element e -> K = (v%4)*2 + (v/4)*16 + g*8 + (e%2), v=e/2, g=lane/16.
// K pairs are memory-contiguous -> 32-bit loads (merge into ds_load_b128).
static __device__ __forceinline__ bf16x16
load_frag_row(const unsigned short* base, int row, int stride, int g) {
    u16x16 t;
    const unsigned short* p = base + (size_t)row * stride;
#pragma unroll
    for (int v = 0; v < 8; ++v) {
        int k = ((v & 3) << 1) + ((v >> 2) << 4) + (g << 3);
        unsigned int u = *(const unsigned int*)(p + k);
        t[2 * v]     = (unsigned short)(u & 0xFFFFu);
        t[2 * v + 1] = (unsigned short)(u >> 16);
    }
    return __builtin_bit_cast(bf16x16, t);
}

static __device__ __forceinline__ f32x8
wmma_bf16(bf16x16 a, bf16x16 b, f32x8 c) {
    return __builtin_amdgcn_wmma_f32_16x16x32_bf16(
        false, a, false, b, (short)0, c, false, false);
}

// Scatter helper: for K index k (0..31) of a 16-bit B fragment, which
// (element, lane-group) slot does it occupy? Inverse of the kmap above.
static __device__ __forceinline__ void
bfrag_slot(int k, int& e, int& g) {
    g = (k >> 3) & 1;
    int v = ((k >> 4) << 2) | ((k & 7) >> 1);
    e = 2 * v + (k & 1);
}

// ---- 16-lane XOR butterfly reductions on VALU (v_permlane16_b32) ----
#if __has_builtin(__builtin_amdgcn_permlane16)
static __device__ __forceinline__ float
lane_xor16(float x, unsigned s0, unsigned s1) {
    unsigned xi = __builtin_bit_cast(unsigned, x);
    xi = __builtin_amdgcn_permlane16(xi, xi, s0, s1, true, false);
    return __builtin_bit_cast(float, xi);
}
static __device__ __forceinline__ float red_max16(float x) {
    x = fmaxf(x, lane_xor16(x, 0x67452301u, 0xEFCDAB89u)); // ^1
    x = fmaxf(x, lane_xor16(x, 0x54761032u, 0xDCFE98BAu)); // ^2
    x = fmaxf(x, lane_xor16(x, 0x32107654u, 0xBA98FEDCu)); // ^4
    x = fmaxf(x, lane_xor16(x, 0xFEDCBA98u, 0x76543210u)); // ^8
    return x;
}
static __device__ __forceinline__ float red_sum16(float x) {
    x += lane_xor16(x, 0x67452301u, 0xEFCDAB89u);
    x += lane_xor16(x, 0x54761032u, 0xDCFE98BAu);
    x += lane_xor16(x, 0x32107654u, 0xBA98FEDCu);
    x += lane_xor16(x, 0xFEDCBA98u, 0x76543210u);
    return x;
}
#else
static __device__ __forceinline__ float red_max16(float x) {
    for (int off = 1; off < 16; off <<= 1) x = fmaxf(x, __shfl_xor(x, off, 16));
    return x;
}
static __device__ __forceinline__ float red_sum16(float x) {
    for (int off = 1; off < 16; off <<= 1) x += __shfl_xor(x, off, 16);
    return x;
}
#endif

// ---------------------------------------------------------------------------
// bf16 WMMA GEMM: C[M,N] = A[M,K] * W[K,N] + bias
// 256 threads = 8 waves (4 along M x 2 along N); block tile 128x128, K-step 32.
// Wave tile 32x64: 2 A-frags x 4 B-frags -> 8 WMMA per K-step.
// B tile staged in LDS in *fragment order* so each B-frag is one contiguous
// 32-byte lane read (2x ds_load_b128) instead of 16 scalar u16 loads.
// ---------------------------------------------------------------------------
template <bool A_F32, bool OUT_F32>
__global__ __launch_bounds__(256) void gemm_wmma(const void* __restrict__ Aptr,
                                                 const float* __restrict__ W,
                                                 const float* __restrict__ bias,
                                                 void* __restrict__ Cptr,
                                                 int M, int N, int K) {
    alignas(32) __shared__ unsigned short aT[128 * 32];      // [m][k]
    alignas(32) __shared__ unsigned short bTf[8 * 32 * 16];  // [ntile][lane][e]

    const int tid  = threadIdx.x;
    const int wave = tid >> 5;
    const int lane = tid & 31;
    const int g    = lane >> 4;
    const int ln   = lane & 15;
    const int wm   = wave & 3;   // 0..3 -> 32 rows each
    const int wn   = wave >> 2;  // 0..1 -> 64 cols each
    const int m0   = blockIdx.x * 128;
    const int n0   = blockIdx.y * 128;

    f32x8 acc[2][4];
#pragma unroll
    for (int t = 0; t < 2; ++t)
#pragma unroll
        for (int nt = 0; nt < 4; ++nt) acc[t][nt] = zero8();

    for (int k0 = 0; k0 < K; k0 += 32) {
        __syncthreads();
        // stage A tile 128x32 (row-major)
        if (A_F32) {
            const float* A = (const float*)Aptr;
            for (int i = tid; i < 128 * 32; i += 256) {
                int r = i >> 5, c = i & 31;
                aT[i] = f2bf(A[(size_t)(m0 + r) * K + k0 + c]);
            }
        } else {
            const unsigned short* A = (const unsigned short*)Aptr;
            for (int i = tid; i < 128 * 32; i += 256) {
                int r = i >> 5, c = i & 31;
                aT[i] = A[(size_t)(m0 + r) * K + k0 + c];
            }
        }
        // stage B tile 32x128, scattered directly into fragment order
        for (int i = tid; i < 32 * 128; i += 256) {
            int k = i >> 7, c = i & 127;
            unsigned short val = f2bf(W[(size_t)(k0 + k) * N + n0 + c]);
            int e, gg;
            bfrag_slot(k, e, gg);
            int dlane = (c & 15) | (gg << 4);
            bTf[(size_t)(c >> 4) * 512 + dlane * 16 + e] = val;
        }
        if (k0 + 32 < K)
            __builtin_prefetch(&W[(size_t)(k0 + 32) * N + n0 + (tid & 127)], 0, 1);
        __syncthreads();

        bf16x16 a[2];
#pragma unroll
        for (int t = 0; t < 2; ++t)
            a[t] = load_frag_row(aT, wm * 32 + t * 16 + ln, 32, g);
#pragma unroll
        for (int nt = 0; nt < 4; ++nt) {
            bf16x16 b = *(const bf16x16*)&bTf[(size_t)(wn * 4 + nt) * 512 + lane * 16];
#pragma unroll
            for (int t = 0; t < 2; ++t)
                acc[t][nt] = wmma_bf16(a[t], b, acc[t][nt]);
        }
    }

    // epilogue: +bias, store
#pragma unroll
    for (int nt = 0; nt < 4; ++nt) {
        const int col = n0 + wn * 64 + nt * 16 + ln;
        const float bv = bias[col];
#pragma unroll
        for (int t = 0; t < 2; ++t) {
#pragma unroll
            for (int r = 0; r < 8; ++r) {
                const int row = m0 + wm * 32 + t * 16 + r + 8 * g;
                const float v = acc[t][nt][r] + bv;
                if (OUT_F32)
                    ((float*)Cptr)[(size_t)row * N + col] = v;
                else
                    ((unsigned short*)Cptr)[(size_t)row * N + col] = f2bf(v);
            }
        }
    }
}

// ---------------------------------------------------------------------------
// Flash attention, causal, scale = 1/sqrt(64) = 0.125.
// qkv: bf16-raw [B*S, 3*D] (Q|K|V), out: bf16-raw [B*S, D].
// Grid: B*H*(S/64), 128 threads = 4 waves; wave w owns 16 q rows.
// K staged row-major (score B-frags are row-type); V staged in fragment
// order (PV B-frags become contiguous 32-byte lane reads).
// ---------------------------------------------------------------------------
#define ATT_B  8
#define ATT_S  1024
#define ATT_H  16
#define ATT_DH 64
#define ATT_LD 3072

__global__ __launch_bounds__(128) void attn_wmma(const unsigned short* __restrict__ qkv,
                                                 unsigned short* __restrict__ out) {
    alignas(32) __shared__ unsigned short kT[32 * 64];      // [key][dh]
    alignas(32) __shared__ unsigned short vTf[4 * 32 * 16]; // [ntile][lane][e]
    alignas(32) __shared__ unsigned short pT[4 * 16 * 32];  // per-wave P [16][32]

    const int bx   = blockIdx.x;
    const int qblk = bx & 15;          // S/64 = 16
    const int bh   = bx >> 4;
    const int h    = bh & 15;
    const int b    = bh >> 4;

    const int tid  = threadIdx.x;
    const int wave = tid >> 5;
    const int lane = tid & 31;
    const int g    = lane >> 4;
    const int ln   = lane & 15;

    const int qbase_blk = qblk * 64;
    const int qbase     = qbase_blk + wave * 16;

    const unsigned short* qp = qkv + (size_t)b * ATT_S * ATT_LD + h * ATT_DH;
    const unsigned short* kp = qp + 1024;
    const unsigned short* vp = qp + 2048;

    // Q fragments: 16 rows x dh=64 -> two 16x32 A fragments (loaded once)
    bf16x16 qa[2];
#pragma unroll
    for (int f = 0; f < 2; ++f)
        qa[f] = load_frag_row(qp + f * 32, qbase + ln, ATT_LD, g);

    f32x8 o[4];
#pragma unroll
    for (int nt = 0; nt < 4; ++nt) o[nt] = zero8();
    float m[8], l[8];
#pragma unroll
    for (int r = 0; r < 8; ++r) { m[r] = -1.0e30f; l[r] = 0.0f; }

    const int nkb = (qbase_blk + 64) >> 5;  // causal key-block limit
    const float inv_sqrt_dh = 0.125f;

    for (int kb = 0; kb < nkb; ++kb) {
        const int k0 = kb * 32;
        __syncthreads();
        // stage K rows (row-major) and V rows (fragment order), 32x64 each
        for (int i = tid; i < 1024; i += 128) {
            int k = i >> 5, c2 = (i & 31) << 1;
            ((unsigned int*)kT)[i] =
                *(const unsigned int*)&kp[(size_t)(k0 + k) * ATT_LD + c2];
            unsigned int u =
                *(const unsigned int*)&vp[(size_t)(k0 + k) * ATT_LD + c2];
            int e, gg;
            bfrag_slot(k, e, gg);
            int dlane = (c2 & 15) | (gg << 4);
            unsigned short* dst = &vTf[(size_t)(c2 >> 4) * 512 + dlane * 16 + e];
            dst[0]  = (unsigned short)(u & 0xFFFFu);
            dst[16] = (unsigned short)(u >> 16);   // next dh col = next lane
        }
        __syncthreads();
        if (kb + 1 < nkb) {
            __builtin_prefetch(&kp[(size_t)(k0 + 32 + (tid & 31)) * ATT_LD], 0, 1);
            __builtin_prefetch(&vp[(size_t)(k0 + 32 + (tid & 31)) * ATT_LD], 0, 1);
        }

        // scores: two 16x16 tiles (32 keys); B column n = key row -> row-type load
        f32x8 sc[2];
#pragma unroll
        for (int kt = 0; kt < 2; ++kt) {
            f32x8 c = zero8();
#pragma unroll
            for (int f = 0; f < 2; ++f) {
                bf16x16 kf = load_frag_row(kT + f * 32, kt * 16 + ln, 64, g);
                c = wmma_bf16(qa[f], kf, c);
            }
            sc[kt] = c;
        }

        // online softmax in C-fragment layout (row = r + 8g, col = ln)
#pragma unroll
        for (int r = 0; r < 8; ++r) {
            const int qr = qbase + r + 8 * g;
            float s0 = sc[0][r] * inv_sqrt_dh;
            float s1 = sc[1][r] * inv_sqrt_dh;
            if (k0 + ln > qr)      s0 = -1.0e9f;
            if (k0 + 16 + ln > qr) s1 = -1.0e9f;

            const float nm = fmaxf(m[r], red_max16(fmaxf(s0, s1)));
            const float rescale = __expf(m[r] - nm);
            m[r] = nm;

            const float p0 = __expf(s0 - nm);
            const float p1 = __expf(s1 - nm);
            l[r] = l[r] * rescale + red_sum16(p0 + p1);
#pragma unroll
            for (int nt = 0; nt < 4; ++nt) o[nt][r] *= rescale;

            pT[wave * 512 + (r + 8 * g) * 32 + ln]      = f2bf(p0);
            pT[wave * 512 + (r + 8 * g) * 32 + 16 + ln] = f2bf(p1);
        }

        // wave-local LDS RAW fence before re-reading P in A-fragment layout
        asm volatile("s_wait_dscnt 0" ::: "memory");

        bf16x16 pa = load_frag_row(pT + wave * 512, ln, 32, g);
#pragma unroll
        for (int nt = 0; nt < 4; ++nt) {
            bf16x16 vb = *(const bf16x16*)&vTf[(size_t)nt * 512 + lane * 16];
            o[nt] = wmma_bf16(pa, vb, o[nt]);
        }
    }

    // normalize + store bf16 [B*S, D]
#pragma unroll
    for (int r = 0; r < 8; ++r) {
        const float invl = 1.0f / l[r];
        const int row = qbase + r + 8 * g;
#pragma unroll
        for (int nt = 0; nt < 4; ++nt)
            out[((size_t)b * ATT_S + row) * 1024 + h * ATT_DH + nt * 16 + ln] =
                f2bf(o[nt][r] * invl);
    }
}

// ---------------------------------------------------------------------------
extern "C" void kernel_launch(void* const* d_in, const int* in_sizes, int n_in,
                              void* d_out, int out_size, void* d_ws, size_t ws_size,
                              hipStream_t stream) {
    const float* x    = (const float*)d_in[0];  // [8,1024,1024]
    const float* Wqkv = (const float*)d_in[1];  // [1024,3072]
    const float* bqkv = (const float*)d_in[2];  // [3072]
    const float* Wout = (const float*)d_in[3];  // [1024,1024]
    const float* bout = (const float*)d_in[4];  // [1024]
    float* out = (float*)d_out;                 // [8,1024,1024]

    const int M = 8 * 1024;  // B*S
    unsigned short* qkv_ws  = (unsigned short*)d_ws;      // 48 MB bf16
    unsigned short* attn_ws = qkv_ws + (size_t)M * 3072;  // 16 MB bf16

    // 1) QKV projection: [8192,1024] x [1024,3072] + b -> bf16 qkv
    dim3 g1(M / 128, 3072 / 128);
    gemm_wmma<true, false><<<g1, 256, 0, stream>>>(
        (const void*)x, Wqkv, bqkv, (void*)qkv_ws, M, 3072, 1024);

    // 2) causal flash attention over 128 (b,h) pairs x 16 q-blocks
    attn_wmma<<<ATT_B * ATT_H * (ATT_S / 64), 128, 0, stream>>>(qkv_ws, attn_ws);

    // 3) output projection: [8192,1024] x [1024,1024] + b -> f32 out
    dim3 g3(M / 128, 1024 / 128);
    gemm_wmma<false, true><<<g3, 256, 0, stream>>>(
        (const void*)attn_ws, Wout, bout, (void*)out, M, 1024, 1024);
}
